// textBiLSTM_90915867721772
// MI455X (gfx1250) — matrix operand order repaired and checked by reading, not compiled, not run on hardware
//
#include <hip/hip_runtime.h>
#include <hip/hip_bf16.h>

typedef __attribute__((ext_vector_type(16))) __bf16 v16bf;
typedef __attribute__((ext_vector_type(8)))  __bf16 v8bf;
typedef __attribute__((ext_vector_type(8)))  float  v8f;

#define E_ 512
#define H_ 1024
#define B_ 256
#define S_ 256
#define K_TOT 1536   // E + H
#define G4H 4096     // 4*H

// ---------- helpers ----------
__device__ __forceinline__ unsigned short f2bf(float f) {
    unsigned u = __float_as_uint(f);
    unsigned r = u + 0x7FFFu + ((u >> 16) & 1u);   // round-to-nearest-even
    return (unsigned short)(r >> 16);
}
__device__ __forceinline__ float bf2f(unsigned short s) {
    return __uint_as_float(((unsigned)s) << 16);
}
__device__ __forceinline__ float sigm(float x) {
    return 1.0f / (1.0f + __expf(-x));
}

union Frag { v16bf v; v8bf h[2]; };

// ---------- zero workspace ----------
__global__ void zero_u32(unsigned* __restrict__ p, long n) {
    long i = (long)blockIdx.x * blockDim.x + threadIdx.x;
    if (i < n) p[i] = 0u;
}

// ---------- embedding gather: x_bf[t][b][e] = bf16(emb[inputs[b][t]][e]) ----------
__global__ void gather_embed(const int* __restrict__ inputs,
                             const float* __restrict__ emb,
                             unsigned* __restrict__ xbf_u32) {
    const long EH = E_ / 2;                       // 2 elems / thread
    long i = (long)blockIdx.x * blockDim.x + threadIdx.x;
    long total = (long)S_ * B_ * EH;
    if (i >= total) return;
    int  t  = (int)(i / (B_ * EH));
    long r  = i - (long)t * (B_ * EH);
    int  b  = (int)(r / EH);
    int  e2 = (int)(r - (long)b * EH);
    int tok = inputs[(long)b * S_ + t];
    const float2 f = *(const float2*)(emb + (size_t)tok * E_ + (size_t)e2 * 2);
    xbf_u32[i] = (unsigned)f2bf(f.x) | ((unsigned)f2bf(f.y) << 16);
}

// ---------- pack weights: Wt[n][k] = bf16( k<512 ? Wx[k][n] : Wh[k-512][n] ) ----------
__global__ void pack_weights(const float* __restrict__ Wx,
                             const float* __restrict__ Wh,
                             unsigned short* __restrict__ Wt) {
    long i = (long)blockIdx.x * blockDim.x + threadIdx.x;
    long total = (long)G4H * K_TOT;
    if (i >= total) return;
    int n = (int)(i / K_TOT);
    int k = (int)(i - (long)n * K_TOT);
    float v = (k < E_) ? Wx[(long)k * G4H + n] : Wh[(long)(k - E_) * G4H + n];
    Wt[i] = f2bf(v);
}

// ---------- one LSTM time step, both directions (blockIdx.z) ----------
// Block: 256 thr = 8 waves, wave grid 4(M) x 2(N); block tile = 64(M) x 32(N of H).
// Each wave: one 16x16 tile at matching (m,n) for all 4 gates -> A-frag reused 4x.
// K loop is 2x unrolled with ping-pong register fragments so the 10 b128 loads
// of slab kk+32 are in flight while the 4 WMMAs of slab kk execute.
__global__ __launch_bounds__(256)
void lstm_step(const unsigned short* __restrict__ xbf,   // [S][B][E] bf16
               const unsigned short* __restrict__ Wt_f,  // [4096][1536] bf16
               const unsigned short* __restrict__ Wt_b,
               const float* __restrict__ bias_f,         // [4096]
               const float* __restrict__ bias_b,
               unsigned short* __restrict__ hbuf,        // [dir][2][B*H] bf16 ping-pong
               float* __restrict__ cbuf,                 // [dir][B*H]
               int t) {
    const int dir = blockIdx.z;
    const unsigned short* Wt   = dir ? Wt_b   : Wt_f;
    const float*          bias = dir ? bias_b : bias_f;
    const int teff = dir ? (S_ - 1 - t) : t;

    const unsigned short* x_t  = xbf + (size_t)teff * B_ * E_;
    const unsigned short* h_in = hbuf + ((size_t)dir * 2 + (t & 1)) * ((size_t)B_ * H_);
    unsigned short*      h_out = hbuf + ((size_t)dir * 2 + ((t + 1) & 1)) * ((size_t)B_ * H_);
    float* c = cbuf + (size_t)dir * B_ * H_;

    const int lane = threadIdx.x & 31;
    const int wid  = threadIdx.x >> 5;
    const int m0 = blockIdx.y * 64 + (wid >> 1) * 16;
    const int n0 = blockIdx.x * 32 + (wid & 1) * 16;

    // A-fragment addressing (16-bit A 16x32): lane half selects K base 0/8,
    // halves 0..7 -> K=kb..kb+7, halves 8..15 -> K=kb+16..kb+23
    const int arow = m0 + (lane & 15);
    const int akb  = (lane >> 4) << 3;     // 0 or 8
    // B-fragment addressing: lane -> column N, lane half selects K base 0/8
    const int bn   = n0 + (lane & 15);
    const int bkb  = (lane >> 4) << 3;     // 0 or 8

    const unsigned short* xbase = x_t  + (size_t)arow * E_ + akb;   // + kk
    const unsigned short* hbase = h_in + (size_t)arow * H_ + akb;   // + (kk - E_)
    const unsigned short* Bbase[4];
#pragma unroll
    for (int g = 0; g < 4; ++g)
        Bbase[g] = Wt + (size_t)(g * H_ + bn) * K_TOT + bkb;        // + kk

    v8f acc[4] = {};                       // gates i, f, o, c

    auto loadA = [&](Frag& f, int kk) {
        const unsigned short* p = (kk < E_) ? (xbase + kk) : (hbase + (kk - E_));
        f.h[0] = *(const v8bf*)(p);
        f.h[1] = *(const v8bf*)(p + 16);
    };
    auto loadB4 = [&](Frag* f4, int kk) {
#pragma unroll
        for (int g = 0; g < 4; ++g) {
            const unsigned short* p = Bbase[g] + kk;
            f4[g].h[0] = *(const v8bf*)(p);
            f4[g].h[1] = *(const v8bf*)(p + 16);
        }
    };

    Frag a0, a1;
    Frag b0[4], b1[4];

    loadA(a0, 0);
    loadB4(b0, 0);

    for (int kk = 0; kk < K_TOT; kk += 64) {
        // prefetch slab kk+32 while computing slab kk
        loadA(a1, kk + 32);
        loadB4(b1, kk + 32);
#pragma unroll
        for (int g = 0; g < 4; ++g)
            acc[g] = __builtin_amdgcn_wmma_f32_16x16x32_bf16(
                         false, a0.v, false, b0[g].v, (short)0, acc[g], false, false);
        // prefetch slab kk+64 while computing slab kk+32
        if (kk + 64 < K_TOT) {
            loadA(a0, kk + 64);
            loadB4(b0, kk + 64);
        }
#pragma unroll
        for (int g = 0; g < 4; ++g)
            acc[g] = __builtin_amdgcn_wmma_f32_16x16x32_bf16(
                         false, a1.v, false, b1[g].v, (short)0, acc[g], false, false);
    }

    // epilogue: C layout lane->N (lane&15), vgpr r -> M = (lane>>4)*8 + r
    const int ncol = n0 + (lane & 15);
    const float bi = bias[0 * H_ + ncol];
    const float bf = bias[1 * H_ + ncol];
    const float bo = bias[2 * H_ + ncol];
    const float bc = bias[3 * H_ + ncol];
    const int mrow0 = m0 + ((lane >> 4) << 3);
#pragma unroll
    for (int r = 0; r < 8; ++r) {
        const size_t idx = (size_t)(mrow0 + r) * H_ + ncol;
        const float gi = acc[0][r] + bi;
        const float gf = acc[1][r] + bf;
        const float go = acc[2][r] + bo;
        const float gc = acc[3][r] + bc;
        const float cn = sigm(gf) * c[idx] + sigm(gi) * tanhf(gc);
        c[idx] = cn;
        h_out[idx] = f2bf(sigm(go) * tanhf(cn));
    }
}

// ---------- head: logits = (h_f + h_b) @ W_hq + b_q ; softmax over 10 ----------
__global__ void head_kernel(const unsigned short* __restrict__ hf,
                            const unsigned short* __restrict__ hb,
                            const float* __restrict__ Whq,   // [H][10]
                            const float* __restrict__ bq,    // [10]
                            float* __restrict__ out) {       // [B][10]
    const int b = blockIdx.x;
    const int tid = threadIdx.x;                 // 64 threads
    __shared__ float part[64][10];
    float s[10];
#pragma unroll
    for (int j = 0; j < 10; ++j) s[j] = 0.0f;
    for (int h = tid; h < H_; h += 64) {
        const float hs = bf2f(hf[(size_t)b * H_ + h]) + bf2f(hb[(size_t)b * H_ + h]);
#pragma unroll
        for (int j = 0; j < 10; ++j) s[j] += hs * Whq[(size_t)h * 10 + j];
    }
#pragma unroll
    for (int j = 0; j < 10; ++j) part[tid][j] = s[j];
    __syncthreads();
    if (tid == 0) {
        float lg[10], mx = -1e30f;
        for (int j = 0; j < 10; ++j) {
            float a = bq[j];
            for (int i = 0; i < 64; ++i) a += part[i][j];
            lg[j] = a;
            mx = fmaxf(mx, a);
        }
        float sum = 0.0f;
        for (int j = 0; j < 10; ++j) { lg[j] = __expf(lg[j] - mx); sum += lg[j]; }
        const float inv = 1.0f / sum;
        for (int j = 0; j < 10; ++j) out[(size_t)b * 10 + j] = lg[j] * inv;
    }
}

// ---------- launch ----------
extern "C" void kernel_launch(void* const* d_in, const int* in_sizes, int n_in,
                              void* d_out, int out_size, void* d_ws, size_t ws_size,
                              hipStream_t stream) {
    const int*   inputs = (const int*)  d_in[0];
    const float* emb    = (const float*)d_in[1];
    const float* Wx_f   = (const float*)d_in[2];
    const float* Wh_f   = (const float*)d_in[3];
    const float* b_f    = (const float*)d_in[4];
    const float* Wx_b   = (const float*)d_in[5];
    const float* Wh_b   = (const float*)d_in[6];
    const float* b_b    = (const float*)d_in[7];
    const float* W_hq   = (const float*)d_in[8];
    const float* b_q    = (const float*)d_in[9];
    float* out = (float*)d_out;

    char* ws = (char*)d_ws;
    size_t off = 0;
    auto alloc = [&](size_t bytes) -> void* {
        void* p = ws + off;
        off += (bytes + 255) & ~(size_t)255;
        return p;
    };
    unsigned short* xbf  = (unsigned short*)alloc((size_t)S_ * B_ * E_ * 2);       // 67 MB
    unsigned short* WtF  = (unsigned short*)alloc((size_t)G4H * K_TOT * 2);        // 12.6 MB
    unsigned short* WtB  = (unsigned short*)alloc((size_t)G4H * K_TOT * 2);        // 12.6 MB
    unsigned short* hbuf = (unsigned short*)alloc((size_t)2 * 2 * B_ * H_ * 2);    // 2 MB
    float*          cbuf = (float*)         alloc((size_t)2 * B_ * H_ * 4);        // 2 MB

    // zero h ping-pong + c state (hbuf and cbuf are contiguous, sizes 256-aligned)
    {
        long n_u32 = ((long)2 * 2 * B_ * H_ * 2 + (long)2 * B_ * H_ * 4) / 4;
        long blocks = (n_u32 + 255) / 256;
        zero_u32<<<dim3((unsigned)blocks), 256, 0, stream>>>((unsigned*)hbuf, n_u32);
    }
    // embedding gather -> bf16 [S,B,E]
    {
        long total = (long)S_ * B_ * (E_ / 2);
        gather_embed<<<dim3((unsigned)((total + 255) / 256)), 256, 0, stream>>>(
            inputs, emb, (unsigned*)xbf);
    }
    // pack [Wx;Wh] -> transposed bf16 [4096][1536], both directions
    {
        long total = (long)G4H * K_TOT;
        unsigned blocks = (unsigned)((total + 255) / 256);
        pack_weights<<<blocks, 256, 0, stream>>>(Wx_f, Wh_f, WtF);
        pack_weights<<<blocks, 256, 0, stream>>>(Wx_b, Wh_b, WtB);
    }
    // recurrence: one launch per step, both directions via gridDim.z
    {
        dim3 grid(H_ / 32, B_ / 64, 2);          // 32 x 4 x 2 = 256 blocks/step
        for (int t = 0; t < S_; ++t) {
            lstm_step<<<grid, 256, 0, stream>>>(xbf, WtF, WtB, b_f, b_b, hbuf, cbuf, t);
        }
    }
    // final h for each direction lands in ping-pong slot 0 (t=255 writes (t+1)&1 == 0)
    const unsigned short* hF = hbuf + (size_t)0 * B_ * H_;
    const unsigned short* hB = hbuf + (size_t)2 * B_ * H_;
    head_kernel<<<B_, 64, 0, stream>>>(hF, hB, W_hq, b_q, out);
}
